// ContrastiveMoCo_29764123361916
// MI455X (gfx1250) — matrix operand order repaired and checked by reading, not compiled, not run on hardware
//
#include <hip/hip_runtime.h>
#include <cmath>

// ---------------------------------------------------------------------------
// MoCo contrastive loss on MI455X (gfx1250, wave32).
//   B=256 H=768 K=65536 L=10
//  - heads: f32 WMMA 16x16x4, fully templated (no exec churn, constant strides)
//  - cos  : bf16 WMMA 16x16x32; fq tile streamed by TDM (tensor_load_to_lds,
//           HW-padded LDS rows), q tile by global_load_async_to_lds_b128;
//           double-buffered, s_wait_tensorcnt / s_wait_asynccnt synchronized
//  - queue scatter handled analytically (no 192MB copy): fixup of B columns
//  - exact topk(neg_min) via 2-pass 16-bit radix select (64K-bin LDS hist)
// ---------------------------------------------------------------------------

#define B_ 256
#define H_ 768
#define K_ 65536
#define L_ 10
#define M_MOM 0.999f
#define TEMP_ 0.07f
#define C_RATE 0.1f

typedef float    v2f   __attribute__((ext_vector_type(2)));
typedef float    v8f   __attribute__((ext_vector_type(8)));
typedef __bf16   v16bf __attribute__((ext_vector_type(16)));
typedef unsigned u32x4 __attribute__((ext_vector_type(4)));
typedef int      i32x4 __attribute__((ext_vector_type(4)));
typedef int      i32x8 __attribute__((ext_vector_type(8)));

// ---- workspace layout (units: floats) ----
constexpr size_t OFF_WK1N = 0;
constexpr size_t OFF_WK2N = OFF_WK1N + (size_t)H_ * H_;
constexpr size_t OFF_BK1N = OFF_WK2N + (size_t)H_ * H_;
constexpr size_t OFF_BK2N = OFF_BK1N + H_;
constexpr size_t OFF_HID  = OFF_BK2N + H_;
constexpr size_t OFF_RAW  = OFF_HID  + (size_t)B_ * H_;
constexpr size_t OFF_KEYS = OFF_RAW  + (size_t)B_ * H_;
constexpr size_t OFF_Q    = OFF_KEYS + (size_t)B_ * H_;
constexpr size_t OFF_LOGC = OFF_Q    + (size_t)B_ * H_;
constexpr size_t OFF_LPOS = OFF_LOGC + (size_t)B_ * L_;
constexpr size_t OFF_LCON = OFF_LPOS + B_;
constexpr size_t OFF_LCLS = OFF_LCON + B_;
constexpr size_t OFF_LQ   = OFF_LCLS + B_;        // K ints
constexpr size_t OFF_NCNT = OFF_LQ   + K_;        // B ints
constexpr size_t OFF_NMIN = OFF_NCNT + B_;        // 1 int
constexpr size_t OFF_HIB  = OFF_NMIN + 1;         // B ints
constexpr size_t OFF_RNK2 = OFF_HIB  + B_;        // B ints
constexpr size_t OFF_RMXU = OFF_RNK2 + B_;        // B uints
constexpr size_t OFF_TU   = OFF_RMXU + B_;        // B uints
constexpr size_t OFF_COS  = (OFF_TU + B_ + 63) & ~(size_t)63;  // B*K floats

__device__ __forceinline__ unsigned f2ord(float v) {
  unsigned u = __float_as_uint(v);
  return (u & 0x80000000u) ? ~u : (u | 0x80000000u);
}
__device__ __forceinline__ float ord2f(unsigned u) {
  unsigned b = (u & 0x80000000u) ? (u ^ 0x80000000u) : ~u;
  return __uint_as_float(b);
}

// async 16B global->LDS copy (per active lane). VDST = LDS byte offset,
// VADDR = 64-bit global address, SADDR = off. Tracked by ASYNCcnt.
__device__ __forceinline__ void async_copy16(void* lds_dst, const void* gsrc) {
  unsigned ldsOff = (unsigned)(uintptr_t)lds_dst;   // LDS aperture: offset in addr[31:0]
  unsigned long long ga = (unsigned long long)(uintptr_t)gsrc;
  asm volatile("global_load_async_to_lds_b128 %0, %1, off"
               :: "v"(ldsOff), "v"(ga) : "memory");
}
__device__ __forceinline__ void wait_async_le2() {
  asm volatile("s_wait_asynccnt 0x2" ::: "memory");
}
__device__ __forceinline__ void wait_async_0() {
  asm volatile("s_wait_asynccnt 0x0" ::: "memory");
}

// TDM: one 2D tile load fq[128 x 32 f32] -> LDS with HW row padding of
// 4 DWORDs after every 32 DWORDs (row stride 36 floats = SB_).
// D# per CDNA5 ISA. Tracked by TENSORcnt.
__device__ __forceinline__ void tdm_load_tileB(const void* gsrc, void* lds_dst) {
  unsigned long long ga = (unsigned long long)(uintptr_t)gsrc;
  unsigned lds = (unsigned)(uintptr_t)lds_dst;
  u32x4 g0;
  g0[0] = 1u;                                      // count=1, user descriptor
  g0[1] = lds;                                     // lds_addr (bytes)
  g0[2] = (unsigned)(ga & 0xffffffffu);            // global_addr[31:0]
  g0[3] = (unsigned)((ga >> 32) & 0x01ffffffu)     // global_addr[56:32]
        | (2u << 30);                              // type = 2 ("image")
  i32x8 g1;
  g1[0] = (int)((2u << 16)                         // data_size = 4B
              | (1u << 20)                         // pad_enable
              | (4u << 22)                         // pad_interval: 32 DWORDs
              | (3u << 25));                       // pad_amount: 4 DWORDs
  g1[1] = (int)(((unsigned)H_ & 0xffffu) << 16);   // tensor_dim0 lo (768)
  g1[2] = (int)(((unsigned)H_ >> 16)               // tensor_dim0 hi
              | (((unsigned)K_ & 0xffffu) << 16)); // tensor_dim1 lo (65536)
  g1[3] = (int)(((unsigned)K_ >> 16)               // tensor_dim1 hi
              | (32u << 16));                      // tile_dim0 = 32
  g1[4] = (int)128u;                               // tile_dim1 = 128 (tile_dim2 = 0)
  g1[5] = (int)H_;                                 // tensor_dim0_stride lo (768)
  g1[6] = 0;                                       // stride hi | dim1_stride lo
  g1[7] = 0;
  i32x4 gz = {0, 0, 0, 0};
#if defined(__clang_major__) && (__clang_major__ >= 23)
  i32x8 gz8 = {0, 0, 0, 0, 0, 0, 0, 0};
  __builtin_amdgcn_tensor_load_to_lds(g0, g1, gz, gz, gz8, 0);
#else
  __builtin_amdgcn_tensor_load_to_lds(g0, g1, gz, gz, 0);
#endif
}

// ---------------------------------------------------------------------------
// 1) momentum update of k-head params
__global__ void k_momentum(const float* Wq1, const float* bq1,
                           const float* Wq2, const float* bq2,
                           const float* Wk1, const float* bk1,
                           const float* Wk2, const float* bk2,
                           float* w1n, float* b1n, float* w2n, float* b2n) {
  int i = blockIdx.x * blockDim.x + threadIdx.x;
  if (i < H_ * H_) {
    w1n[i] = M_MOM * Wk1[i] + (1.0f - M_MOM) * Wq1[i];
    w2n[i] = M_MOM * Wk2[i] + (1.0f - M_MOM) * Wq2[i];
  }
  if (i < H_) {
    b1n[i] = M_MOM * bk1[i] + (1.0f - M_MOM) * bq1[i];
    b2n[i] = M_MOM * bk2[i] + (1.0f - M_MOM) * bq2[i];
  }
}

// ---------------------------------------------------------------------------
// 2) head GEMM: out[M,ND] = act(A[M,KD] @ W[KD,ND] + bias), f32 WMMA 16x16x4.
//    All dims compile-time; B-column clamp instead of exec-diverging guard
//    (garbage B columns >= ND only feed D columns >= ND, which are not stored).
template<int KD, int ND, int ACT>
__global__ void k_gemm_head(const float* __restrict__ A,
                            const float* __restrict__ W,
                            const float* __restrict__ bias,
                            float* __restrict__ out) {
  constexpr int NTN = (ND + 15) / 16;
  int lane  = threadIdx.x & 31;
  int wave  = threadIdx.x >> 5;
  int tile  = blockIdx.x * 4 + wave;          // blockDim.x == 128
  int tm    = tile / NTN, tn = tile % NTN;
  int m0    = tm << 4,  n0 = tn << 4;
  int mrow  = lane & 15;
  int khalf = lane >> 4;                      // A/B lanes hold K = khalf*2 + {0,1}
  int ncol  = n0 + mrow;
  int ncl   = ncol < ND ? ncol : ND - 1;      // clamped column (no divergence)

  const float* ap = A + (size_t)(m0 + mrow) * KD + khalf * 2;
  const float* wp = W + (size_t)(khalf * 2) * ND + ncl;

  v8f acc = {};
#pragma unroll 4
  for (int k0 = 0; k0 < KD; k0 += 4) {
    v2f a; a.x = ap[0]; a.y = ap[1];
    v2f b; b.x = wp[0]; b.y = wp[ND];
    acc = __builtin_amdgcn_wmma_f32_16x16x4_f32(false, a, false, b,
                                                (short)0, acc, false, false);
    ap += 4; wp += 4 * ND;
  }
  if (ncol < ND) {
    float bi = bias[ncol];
#pragma unroll
    for (int r = 0; r < 8; r++) {
      int m = m0 + r + khalf * 8;
      float v = acc[r] + bi;
      if (ACT) v = tanhf(v);
      out[(size_t)m * ND + ncol] = v;
    }
  }
}

// ---------------------------------------------------------------------------
// 3) row-wise l2 normalization (block per row)
__global__ void k_l2norm(const float* __restrict__ in, float* __restrict__ out,
                         int cols) {
  __shared__ float red[256];
  int row = blockIdx.x, tid = threadIdx.x;
  const float* r = in + (size_t)row * cols;
  float s = 0.f;
  for (int c = tid; c < cols; c += 256) { float v = r[c]; s += v * v; }
  red[tid] = s; __syncthreads();
  for (int o = 128; o > 0; o >>= 1) {
    if (tid < o) red[tid] += red[tid + o];
    __syncthreads();
  }
  float inv = rsqrtf(red[0]);
  for (int c = tid; c < cols; c += 256) out[(size_t)row * cols + c] = r[c] * inv;
}

// ---------------------------------------------------------------------------
// 4) updated label queue (wrap-around scatter window handled analytically)
__global__ void k_lqnew(const int* __restrict__ label_queue,
                        const int* __restrict__ labels,
                        const int* __restrict__ ptrp,
                        int* __restrict__ lq) {
  int k = blockIdx.x * blockDim.x + threadIdx.x;
  int d = (k - ptrp[0]) & (K_ - 1);
  lq[k] = (d < B_) ? labels[d] : label_queue[k];
}

// 5) per-row negative counts + global min
__global__ void k_negstats(const int* __restrict__ lq, const int* __restrict__ labels,
                           int* __restrict__ ncnt) {
  __shared__ int red[256];
  int row = blockIdx.x, tid = threadIdx.x;
  int lab = labels[row];
  int c = 0;
  for (int k = tid; k < K_; k += 256) c += (lq[k] != lab) ? 1 : 0;
  red[tid] = c; __syncthreads();
  for (int o = 128; o > 0; o >>= 1) {
    if (tid < o) red[tid] += red[tid + o];
    __syncthreads();
  }
  if (tid == 0) ncnt[row] = red[0];
}

__global__ void k_minred(const int* __restrict__ ncnt, int* __restrict__ nmin) {
  __shared__ int red[256];
  int tid = threadIdx.x;
  int m = 0x7fffffff;
  for (int i = tid; i < B_; i += 256) m = min(m, ncnt[i]);
  red[tid] = m; __syncthreads();
  for (int o = 128; o > 0; o >>= 1) {
    if (tid < o) red[tid] = min(red[tid], red[tid + o]);
    __syncthreads();
  }
  if (tid == 0) *nmin = red[0];
}

// ---------------------------------------------------------------------------
// 6) cos[B,K] = q[B,H] @ fq[K,H]^T, bf16 WMMA 16x16x32.
//    Block 256 thr (8 waves) -> 64M x 128N tile; K staged in 32-chunks.
//    fq tile: one TDM tensor_load_to_lds per stage (HW pad -> stride 36).
//    q  tile: per-lane global_load_async_to_lds_b128 (2 per thread).
//    Double-buffered; s_wait_tensorcnt + s_wait_asynccnt + barrier per stage.
#define SB_ 36
#define KC_ 32
__global__ void k_cosgemm(const float* __restrict__ q,
                          const float* __restrict__ fq,
                          float* __restrict__ cos) {
  __shared__ float ldsB[2][128 * SB_];   // fq tile: 128 rows x 32 K (+4 pad)
  __shared__ float ldsA[2][64  * SB_];   // q  tile:  64 rows x 32 K (+4 pad)

  int t     = threadIdx.x;
  int lane  = t & 31;
  int wave  = t >> 5;
  int mrow  = lane & 15;
  int khalf = lane >> 4;
  int m0    = blockIdx.x * 64;
  int nblk  = blockIdx.y * 128;
  bool tdm_wave = (__builtin_amdgcn_readfirstlane(wave) == 0); // provably uniform

  // A tile async fills: 2 x b128 per thread (64 rows x 8 chunks = 512)
  auto issueA = [&](int s, int buf) {
    int kk = s * KC_;
#pragma unroll
    for (int i = 0; i < 2; i++) {
      int ca = t + 256 * i;
      int row = ca >> 3, ch = ca & 7;
      async_copy16(&ldsA[buf][row * SB_ + ch * 4],
                   q + (size_t)(m0 + row) * H_ + kk + ch * 4);
    }
  };
  // B tile: single TDM descriptor load (issued by wave 0 only)
  auto issueB = [&](int s, int buf) {
    if (tdm_wave)
      tdm_load_tileB(fq + (size_t)nblk * H_ + s * KC_, &ldsB[buf][0]);
  };

  v8f acc[4] = {};
  constexpr int S = H_ / KC_;              // 24 stages
  issueA(0, 0); issueB(0, 0);

  for (int s = 0; s < S; ++s) {
    int buf = s & 1;
    if (s + 1 < S) {
      issueA(s + 1, buf ^ 1); issueB(s + 1, buf ^ 1);
      wait_async_le2();                            // stage-s A data landed
      __builtin_amdgcn_s_wait_tensorcnt((short)1); // stage-s B data landed
    } else {
      wait_async_0();
      __builtin_amdgcn_s_wait_tensorcnt((short)0);
    }
    __syncthreads();                       // all waves' deposits visible

    // B fragment 32x16: lane col = wave*16+mrow, K = khalf*16 + 0..15
    const float* bp = &ldsB[buf][(wave * 16 + mrow) * SB_ + khalf * 16];
    v16bf b;
#pragma unroll
    for (int j = 0; j < 16; j++) b[j] = (__bf16)bp[j];

#pragma unroll
    for (int mt = 0; mt < 4; mt++) {
      // A fragment 16x32: lane row = mt*16+mrow, K = khalf*16 + 0..15
      const float* ap = &ldsA[buf][(mt * 16 + mrow) * SB_ + khalf * 16];
      v16bf a;
#pragma unroll
      for (int j = 0; j < 8; j++) a[j] = (__bf16)ap[j];
#pragma unroll
      for (int j = 0; j < 8; j++) a[8 + j] = (__bf16)ap[8 + j];
      acc[mt] = __builtin_amdgcn_wmma_f32_16x16x32_bf16(false, a, false, b,
                                                        (short)0, acc[mt],
                                                        false, false);
    }
    __syncthreads();                       // protect buf before next overwrite
  }

  int nrow = nblk + wave * 16 + mrow;
#pragma unroll
  for (int mt = 0; mt < 4; mt++)
#pragma unroll
    for (int r = 0; r < 8; r++) {
      int m = m0 + mt * 16 + r + khalf * 8;
      cos[(size_t)m * K_ + nrow] = acc[mt][r];
    }
}

// ---------------------------------------------------------------------------
// 7) fixup: overwrite the B scattered queue columns with exact f32 dots
__global__ void k_fixup(const float* __restrict__ q, const float* __restrict__ keys,
                        const int* __restrict__ ptrp,
                        float* __restrict__ cos, float* __restrict__ l_pos) {
  __shared__ float qr[H_];
  int i = blockIdx.x, j = threadIdx.x;
  for (int h = j; h < H_; h += 256) qr[h] = q[(size_t)i * H_ + h];
  __syncthreads();
  const float* kp = keys + (size_t)j * H_;
  float s = 0.f;
  for (int h = 0; h < H_; h++) s += qr[h] * kp[h];
  int c = (ptrp[0] + j) & (K_ - 1);
  cos[(size_t)i * K_ + c] = s;
  if (j == i) l_pos[i] = s;
}

// ---------------------------------------------------------------------------
// 8) exact radix select of the neg_min-th largest masked value per row.
__global__ void k_select(const float* __restrict__ cos, const int* __restrict__ lq,
                         const int* __restrict__ labels, const int* __restrict__ nminp,
                         int pass, int* __restrict__ hi_bin, int* __restrict__ rank2,
                         unsigned* __restrict__ rmax_u, unsigned* __restrict__ t_u) {
  extern __shared__ unsigned hist[];   // 65536 bins (256 KB dynamic LDS)
  __shared__ unsigned part[256];
  __shared__ unsigned mred[256];
  int row = blockIdx.x, tid = threadIdx.x;
  for (int b = tid; b < 65536; b += 256) hist[b] = 0u;
  __syncthreads();

  int lab = labels[row];
  int hb  = pass ? hi_bin[row] : -1;
  const float* cr = cos + (size_t)row * K_;
  unsigned mx = 0u;
  for (int k = tid; k < K_; k += 256) {
    __builtin_prefetch((const void*)(cr + k + 2048), 0, 0);
    if (lq[k] != lab) {
      unsigned u = f2ord(cr[k]);
      if (pass == 0) { atomicAdd(&hist[u >> 16], 1u); mx = max(mx, u); }
      else if ((int)(u >> 16) == hb) atomicAdd(&hist[u & 0xffffu], 1u);
    }
  }
  __syncthreads();

  unsigned ps = 0u;
  for (int b = 0; b < 256; b++) ps += hist[tid * 256 + b];
  part[tid] = ps;
  mred[tid] = mx;
  __syncthreads();

  if (tid == 0) {
    unsigned target = pass == 0 ? (unsigned)*nminp : (unsigned)rank2[row];
    unsigned cum = 0u; int c = 255;
    for (; c > 0; c--) { if (cum + part[c] >= target) break; cum += part[c]; }
    int bin = 0;
    for (int b = 255; b >= 0; b--) {
      unsigned h = hist[c * 256 + b];
      if (cum + h >= target) { bin = c * 256 + b; break; }
      cum += h;
    }
    if (pass == 0) {
      hi_bin[row] = bin;
      rank2[row]  = (int)(target - cum);      // rank within bucket, from top
      unsigned mm = 0u;
      for (int x = 0; x < 256; x++) mm = max(mm, mred[x]);
      rmax_u[row] = mm;
    } else {
      t_u[row] = ((unsigned)hi_bin[row] << 16) | (unsigned)bin;
    }
  }
}

// ---------------------------------------------------------------------------
// 9) classifier CE rows (L=10) — one thread per row
__global__ void k_loss_cls(const float* __restrict__ logits,
                           const int* __restrict__ labels,
                           float* __restrict__ out_row) {
  int i = threadIdx.x;
  const float* r = logits + (size_t)i * L_;
  float m = r[0];
  for (int c = 1; c < L_; c++) m = fmaxf(m, r[c]);
  float s = 0.f;
  for (int c = 0; c < L_; c++) s += expf(r[c] - m);
  out_row[i] = m + logf(s) - r[labels[i]];
}

// 10) contrastive CE rows: stable logsumexp over {l_pos, top-neg_min negatives}
__global__ void k_loss_con(const float* __restrict__ cos, const int* __restrict__ lq,
                           const int* __restrict__ labels,
                           const float* __restrict__ l_pos,
                           const unsigned* __restrict__ t_u,
                           const unsigned* __restrict__ rmax_u,
                           const int* __restrict__ nminp,
                           float* __restrict__ out_row) {
  __shared__ float ss[256];
  __shared__ int   sc[256];
  int i = blockIdx.x, tid = threadIdx.x;
  int lab = labels[i];
  int nm  = *nminp;
  unsigned tu = t_u[i];
  float tf   = ord2f(tu);
  float lp   = l_pos[i];
  float zmax = fmaxf(ord2f(rmax_u[i]), lp);
  const float rT = 1.0f / TEMP_;
  const float* cr = cos + (size_t)i * K_;
  float s = 0.f; int cgt = 0;
  for (int k = tid; k < K_; k += 256) {
    __builtin_prefetch((const void*)(cr + k + 2048), 0, 0);
    if (lq[k] != lab) {
      float v = cr[k];
      if (f2ord(v) > tu) { s += expf((v - zmax) * rT); cgt++; }
    }
  }
  ss[tid] = s; sc[tid] = cgt; __syncthreads();
  for (int o = 128; o > 0; o >>= 1) {
    if (tid < o) { ss[tid] += ss[tid + o]; sc[tid] += sc[tid + o]; }
    __syncthreads();
  }
  if (tid == 0) {
    float total = ss[0] + (float)(nm - sc[0]) * expf((tf - zmax) * rT)
                        + expf((lp - zmax) * rT);
    out_row[i] = zmax * rT + logf(total) - lp * rT;
  }
}

// 11) final combine
__global__ void k_final(const float* __restrict__ lcon, const float* __restrict__ lcls,
                        const int* __restrict__ nminp, float* __restrict__ out) {
  __shared__ float a[256], b[256];
  int tid = threadIdx.x;
  a[tid] = lcon[tid]; b[tid] = lcls[tid];   // B == 256 == blockDim
  __syncthreads();
  for (int o = 128; o > 0; o >>= 1) {
    if (tid < o) { a[tid] += a[tid + o]; b[tid] += b[tid + o]; }
    __syncthreads();
  }
  if (tid == 0) {
    float mc = a[0] / (float)B_, mcls = b[0] / (float)B_;
    out[0] = (*nminp > 0) ? (C_RATE * mc + (1.0f - C_RATE) * mcls) : mcls;
  }
}

// ---------------------------------------------------------------------------
extern "C" void kernel_launch(void* const* d_in, const int* in_sizes, int n_in,
                              void* d_out, int out_size, void* d_ws, size_t ws_size,
                              hipStream_t stream) {
  const float* pooled_q = (const float*)d_in[0];
  const float* pooled_p = (const float*)d_in[1];
  const int*   labels   = (const int*)d_in[2];
  const int*   lqueue   = (const int*)d_in[3];
  const float* fqueue   = (const float*)d_in[4];
  const float* Wq1 = (const float*)d_in[5];  const float* bq1 = (const float*)d_in[6];
  const float* Wq2 = (const float*)d_in[7];  const float* bq2 = (const float*)d_in[8];
  const float* Wk1 = (const float*)d_in[9];  const float* bk1 = (const float*)d_in[10];
  const float* Wk2 = (const float*)d_in[11]; const float* bk2 = (const float*)d_in[12];
  const float* Wc1 = (const float*)d_in[13]; const float* bc1 = (const float*)d_in[14];
  const float* Wc2 = (const float*)d_in[15]; const float* bc2 = (const float*)d_in[16];
  const int*   ptrp = (const int*)d_in[17];

  float* ws  = (float*)d_ws;
  float* out = (float*)d_out;

  float* w1n  = ws + OFF_WK1N;  float* w2n  = ws + OFF_WK2N;
  float* b1n  = ws + OFF_BK1N;  float* b2n  = ws + OFF_BK2N;
  float* hid  = ws + OFF_HID;   float* raw  = ws + OFF_RAW;
  float* keys = ws + OFF_KEYS;  float* q    = ws + OFF_Q;
  float* logc = ws + OFF_LOGC;  float* lpos = ws + OFF_LPOS;
  float* lcon = ws + OFF_LCON;  float* lcls = ws + OFF_LCLS;
  int*   lq   = (int*)(ws + OFF_LQ);
  int*   ncnt = (int*)(ws + OFF_NCNT);
  int*   nmin = (int*)(ws + OFF_NMIN);
  int*   hib  = (int*)(ws + OFF_HIB);
  int*   rnk2 = (int*)(ws + OFF_RNK2);
  unsigned* rmxu = (unsigned*)(ws + OFF_RMXU);
  unsigned* tu   = (unsigned*)(ws + OFF_TU);
  float* cos  = ws + OFF_COS;

  // 1) momentum update
  k_momentum<<<(H_ * H_ + 255) / 256, 256, 0, stream>>>(
      Wq1, bq1, Wq2, bq2, Wk1, bk1, Wk2, bk2, w1n, b1n, w2n, b2n);

  // head GEMMs: 4 waves/block, exact tile grids (768 tiles -> 192 blocks; 16 -> 4)
  // 2) key head
  k_gemm_head<H_, H_, 1><<<192, 128, 0, stream>>>(pooled_p, w1n, b1n, hid);
  k_gemm_head<H_, H_, 0><<<192, 128, 0, stream>>>(hid, w2n, b2n, raw);
  k_l2norm<<<B_, 256, 0, stream>>>(raw, keys, H_);

  // 3) query head
  k_gemm_head<H_, H_, 1><<<192, 128, 0, stream>>>(pooled_q, Wq1, bq1, hid);
  k_gemm_head<H_, H_, 0><<<192, 128, 0, stream>>>(hid, Wq2, bq2, raw);
  k_l2norm<<<B_, 256, 0, stream>>>(raw, q, H_);

  // 4) classifier head
  k_gemm_head<H_, H_, 1><<<192, 128, 0, stream>>>(pooled_q, Wc1, bc1, hid);
  k_gemm_head<H_, L_, 0><<<4, 128, 0, stream>>>(hid, Wc2, bc2, logc);

  // 5) queue label scatter + negative stats
  k_lqnew<<<K_ / 256, 256, 0, stream>>>(lqueue, labels, ptrp, lq);
  k_negstats<<<B_, 256, 0, stream>>>(lq, labels, ncnt);
  k_minred<<<1, 256, 0, stream>>>(ncnt, nmin);

  // 6) big cosine-sim GEMM (bf16 WMMA + TDM/async LDS staging) + column fixup
  k_cosgemm<<<dim3(B_ / 64, K_ / 128), 256, 0, stream>>>(q, fqueue, cos);
  k_fixup<<<B_, 256, 0, stream>>>(q, keys, ptrp, cos, lpos);

  // 7) exact top-neg_min threshold per row (two radix passes)
  size_t histBytes = 65536 * sizeof(unsigned);
  k_select<<<B_, 256, histBytes, stream>>>(cos, lq, labels, nmin, 0, hib, rnk2, rmxu, tu);
  k_select<<<B_, 256, histBytes, stream>>>(cos, lq, labels, nmin, 1, hib, rnk2, rmxu, tu);

  // 8) losses
  k_loss_cls<<<1, B_, 0, stream>>>(logc, labels, lcls);
  k_loss_con<<<B_, 256, 0, stream>>>(cos, lq, labels, lpos, tu, rmxu, nmin, lcon);
  k_final<<<1, B_, 0, stream>>>(lcon, lcls, nmin, out);
}
